// CNN_LeNetSym_79714593013879
// MI455X (gfx1250) — compile-verified, matrix operand order checked by reading, not run on hardware
//
#include <hip/hip_runtime.h>
#include <hip/hip_bf16.h>

#define NC 512
#define HP 516   // per-thread histogram bytes: 129 dwords, 129 % 64 == 1 -> bank-rotated, conflict-free scan
#define CT 64    // threads per conv block (LDS: 64*516 + image < 40KB)

typedef __attribute__((ext_vector_type(16))) _Float16 v16h;
typedef __attribute__((ext_vector_type(8)))  float    v8f;

// ---------------- Stage 1: nearest-centroid quantization ----------------
__global__ void k_quant(const float* __restrict__ x, const float* __restrict__ cent,
                        int* __restrict__ sym, int npix) {
    __shared__ float sc[NC];
    for (int i = threadIdx.x; i < NC; i += blockDim.x) sc[i] = cent[i];
    __syncthreads();
    int idx = blockIdx.x * blockDim.x + threadIdx.x;
    if (idx >= npix) return;
    float v = x[idx];
    float bd = fabsf(v - sc[0]);
    int bi = 0;
#pragma unroll 4
    for (int c = 1; c < NC; ++c) {
        float d = fabsf(v - sc[c]);
        if (d < bd) { bd = d; bi = c; }   // strict '<' keeps first index on tie (argmin semantics)
    }
    sym[idx] = bi;
}

// --------- counting-sort + sorted left-fold through add_lut (shared helper) ---------
__device__ __forceinline__ int hist_fold(unsigned char* myh, const int* __restrict__ add_lut) {
    const unsigned* h32 = (const unsigned*)myh;
    int carry = -1;
    for (int w = 0; w < NC / 4; ++w) {
        unsigned d = h32[w];
        if (d == 0u) continue;
#pragma unroll
        for (int b = 0; b < 4; ++b) {
            int cnt = (int)((d >> (8 * b)) & 0xffu);
            int v = w * 4 + b;
            while (cnt-- > 0) carry = (carry < 0) ? v : add_lut[v * NC + carry];
        }
    }
    return carry;
}

// ---------------- Stage 2: symbolic conv1 (28x28 -> 12x12x8) + relu_lut ----------------
__global__ void k_conv1(const int* __restrict__ sym, const int* __restrict__ conv_lut,
                        const int* __restrict__ add_lut, const int* __restrict__ relu_lut,
                        const int* __restrict__ c1w, int* __restrict__ h1) {
    __shared__ unsigned char hist[CT * HP];
    __shared__ int simg[784];
    int img = blockIdx.x / 18, chunk = blockIdx.x % 18;
    const int* sp = sym + (size_t)img * 784;
    for (int i = threadIdx.x; i < 784; i += blockDim.x) simg[i] = sp[i];
    unsigned* h32 = (unsigned*)(hist + threadIdx.x * HP);
#pragma unroll
    for (int w = 0; w < HP / 4; ++w) h32[w] = 0u;
    if (threadIdx.x == 0) __builtin_prefetch(conv_lut, 0, 3);
    __syncthreads();

    int task = chunk * CT + threadIdx.x;           // 0..1151
    int win = task >> 3, f = task & 7;             // 144 windows x 8 filters
    int wy = win / 12, wx = win % 12;
    unsigned char* myh = hist + threadIdx.x * HP;
#pragma unroll
    for (int kh = 0; kh < 5; ++kh)
#pragma unroll
        for (int kw = 0; kw < 5; ++kw) {
            int s  = simg[(wy * 2 + kh) * 28 + (wx * 2 + kw)];
            int wv = c1w[(kh * 5 + kw) * 8 + f];   // c1_weights[(25,8)] row-major
            myh[conv_lut[s * NC + wv]]++;
        }
    int carry = hist_fold(myh, add_lut);
    h1[(size_t)img * 1152 + win * 8 + f] = relu_lut[carry];
}

// ---------------- Stage 3: symbolic conv2 (12x12x8 -> 4x4x16) + relu + centroid -> f16 features ----------------
__global__ void k_conv2(const int* __restrict__ h1, const int* __restrict__ conv_lut,
                        const int* __restrict__ add_lut, const int* __restrict__ relu_lut,
                        const int* __restrict__ c2w, const float* __restrict__ cent,
                        _Float16* __restrict__ vfeat) {
    __shared__ unsigned char hist[CT * HP];
    __shared__ int h1s[1152];
    int img = blockIdx.x >> 2, chunk = blockIdx.x & 3;
    const int* hp = h1 + (size_t)img * 1152;
    for (int i = threadIdx.x; i < 1152; i += blockDim.x) h1s[i] = hp[i];
    unsigned* h32 = (unsigned*)(hist + threadIdx.x * HP);
#pragma unroll
    for (int w = 0; w < HP / 4; ++w) h32[w] = 0u;
    if (threadIdx.x == 0) __builtin_prefetch(add_lut, 0, 3);
    __syncthreads();

    int task = chunk * CT + threadIdx.x;           // 0..255
    int win = task >> 4, f = task & 15;            // 16 windows x 16 filters
    int wy = win >> 2, wx = win & 3;
    unsigned char* myh = hist + threadIdx.x * HP;
    for (int c = 0; c < 8; ++c)                    // window flatten order: (C, kh, kw)
        for (int r = 0; r < 5; ++r)
#pragma unroll
            for (int q = 0; q < 5; ++q) {
                int s  = h1s[((wy * 2 + r) * 12 + (wx * 2 + q)) * 8 + c];
                int wv = c2w[(c * 25 + r * 5 + q) * 16 + f];  // c2_weights[(200,16)]
                myh[conv_lut[s * NC + wv]]++;
            }
    int carry = hist_fold(myh, add_lut);
    int s2 = relu_lut[carry];
    // v = centroid[h], transpose (C,H,W) flatten: index = f*16 + win
    vfeat[(size_t)img * 256 + f * 16 + win] = (_Float16)cent[s2];
}

// ---------------- Weight conversion to f16 (W3 zero-padded 10->16 rows) ----------------
__global__ void k_cvtw(const float* __restrict__ W1, const float* __restrict__ W2,
                       const float* __restrict__ W3, _Float16* __restrict__ w1h,
                       _Float16* __restrict__ w2h, _Float16* __restrict__ w3h) {
    int i = blockIdx.x * blockDim.x + threadIdx.x;
    if (i < 128 * 256) w1h[i] = (_Float16)W1[i];
    if (i < 64 * 128)  w2h[i] = (_Float16)W2[i];
    if (i < 16 * 64) {
        int r = i / 64, c = i % 64;
        w3h[i] = (r < 10) ? (_Float16)W3[r * 64 + c] : (_Float16)0.0f;
    }
}

// ---------------- WMMA GEMM: C[M,N] = act(A[M,K] @ W[N,K]^T), one 16x16 tile per wave ----------------
__global__ void k_gemm16(const _Float16* __restrict__ A, const _Float16* __restrict__ W,
                         _Float16* __restrict__ outH, float* __restrict__ outF,
                         int M, int N, int K, int do_relu) {
    int wave = threadIdx.x >> 5;
    int lane = threadIdx.x & 31;
    int tile = blockIdx.x * 4 + wave;
    int ntn  = N >> 4;
    int tn   = tile % ntn;
    int tm   = tile / ntn;
    if (tm >= (M >> 4)) return;                    // wave-uniform guard (EXEC all-ones for WMMA)

    int l16 = lane & 15;
    int khi = (lane & 16) ? 8 : 0;
    int kb[8];
#pragma unroll
    for (int j = 0; j < 8; ++j) kb[j] = ((j & 4) ? 16 : 0) + khi + ((j & 3) << 1);

    const _Float16* arow = A + (size_t)(tm * 16 + l16) * K;  // A: lane = M row
    const _Float16* brow = W + (size_t)(tn * 16 + l16) * K;  // B[k][n] = W[n][k]: lane = N col
    v8f c = {};
    for (int kk = 0; kk < K; kk += 32) {
        v16h a, b;
#pragma unroll
        for (int j = 0; j < 8; ++j) {
            int k0 = kk + kb[j];
            a[2 * j]     = arow[k0];
            a[2 * j + 1] = arow[k0 + 1];
            b[2 * j]     = brow[k0];
            b[2 * j + 1] = brow[k0 + 1];
        }
        c = __builtin_amdgcn_wmma_f32_16x16x32_f16(false, a, false, b, (short)0, c, false, false);
    }
    int n = tn * 16 + l16;
    int mbase = tm * 16 + ((lane & 16) ? 8 : 0);
#pragma unroll
    for (int r = 0; r < 8; ++r) {
        float v = c[r];
        if (do_relu) v = v > 0.f ? v : 0.f;
        int m = mbase + r;
        if (outH) outH[(size_t)m * N + n] = (_Float16)v;
        if (outF) outF[(size_t)m * N + n] = v;
    }
}

// ---------------- Softmax over 10 valid logits ----------------
__global__ void k_softmax(const float* __restrict__ logits, float* __restrict__ out, int B) {
    int r = blockIdx.x * blockDim.x + threadIdx.x;
    if (r >= B) return;
    float v[10], mx = -1e30f;
#pragma unroll
    for (int i = 0; i < 10; ++i) { v[i] = logits[r * 16 + i]; mx = fmaxf(mx, v[i]); }
    float s = 0.f;
#pragma unroll
    for (int i = 0; i < 10; ++i) { v[i] = expf(v[i] - mx); s += v[i]; }
    float inv = 1.f / s;
#pragma unroll
    for (int i = 0; i < 10; ++i) out[r * 10 + i] = v[i] * inv;
}

extern "C" void kernel_launch(void* const* d_in, const int* in_sizes, int n_in,
                              void* d_out, int out_size, void* d_ws, size_t ws_size,
                              hipStream_t stream) {
    const float* x        = (const float*)d_in[0];
    const float* cent     = (const float*)d_in[1];
    const float* W1       = (const float*)d_in[2];
    const float* W2       = (const float*)d_in[3];
    const float* W3       = (const float*)d_in[4];
    const int*   conv_lut = (const int*)d_in[5];
    const int*   add_lut  = (const int*)d_in[6];
    const int*   relu_lut = (const int*)d_in[7];
    const int*   c1w      = (const int*)d_in[8];
    const int*   c2w      = (const int*)d_in[9];
    float* out = (float*)d_out;

    int B = in_sizes[0] / 784;

    char* ws = (char*)d_ws;
    size_t off = 0;
    auto alloc = [&](size_t bytes) -> void* {
        off = (off + 255) & ~(size_t)255;
        void* p = ws + off;
        off += bytes;
        return p;
    };
    int*       sym = (int*)alloc((size_t)B * 784 * 4);
    int*       h1  = (int*)alloc((size_t)B * 1152 * 4);
    _Float16*  vf  = (_Float16*)alloc((size_t)B * 256 * 2);
    _Float16*  a1  = (_Float16*)alloc((size_t)B * 128 * 2);
    _Float16*  a2  = (_Float16*)alloc((size_t)B * 64 * 2);
    float*     lg  = (float*)alloc((size_t)B * 16 * 4);
    _Float16*  w1h = (_Float16*)alloc(128 * 256 * 2);
    _Float16*  w2h = (_Float16*)alloc(64 * 128 * 2);
    _Float16*  w3h = (_Float16*)alloc(16 * 64 * 2);

    k_quant<<<(B * 784 + 255) / 256, 256, 0, stream>>>(x, cent, sym, B * 784);
    k_cvtw<<<128, 256, 0, stream>>>(W1, W2, W3, w1h, w2h, w3h);
    k_conv1<<<B * 18, CT, 0, stream>>>(sym, conv_lut, add_lut, relu_lut, c1w, h1);
    k_conv2<<<B * 4, CT, 0, stream>>>(h1, conv_lut, add_lut, relu_lut, c2w, cent, vf);

    int tilesM = B >> 4;
    k_gemm16<<<(tilesM * 8 + 3) / 4, 128, 0, stream>>>(vf, w1h, a1, nullptr, B, 128, 256, 1);
    k_gemm16<<<(tilesM * 4 + 3) / 4, 128, 0, stream>>>(a1, w2h, a2, nullptr, B, 64, 128, 1);
    k_gemm16<<<(tilesM * 1 + 3) / 4, 128, 0, stream>>>(a2, w3h, nullptr, lg, B, 16, 64, 0);

    k_softmax<<<(B + 255) / 256, 256, 0, stream>>>(lg, out, B);
}
